// Decoder_53523882443003
// MI455X (gfx1250) — matrix operand order repaired and checked by reading, not compiled, not run on hardware
//
#include <hip/hip_runtime.h>
#include <hip/hip_bf16.h>

#define B_      64
#define L_      196
#define DIMG    2048
#define RNN_    1024
#define EMB_    512
#define TSTEPS  17
#define VOUT    10003
#define VOUT_PAD 10016
#define START_TOK 10001

typedef __attribute__((ext_vector_type(16))) __bf16 v16bf;
typedef __attribute__((ext_vector_type(8)))  float  v8f;

__device__ __forceinline__ unsigned short f2bf(float f) {
    unsigned int u = __builtin_bit_cast(unsigned int, f);
    unsigned int r = u + 0x7fffu + ((u >> 16) & 1u);   // round-to-nearest-even
    return (unsigned short)(r >> 16);
}

union Frag16 { v16bf v; unsigned short s[16]; unsigned int u[8]; };

// ---- fp32 -> bf16 flat convert (activations) --------------------------------
__global__ void k_convA(const float* __restrict__ in, unsigned short* __restrict__ out,
                        int total) {
    int idx = blockIdx.x * blockDim.x + threadIdx.x;
    if (idx < total) out[idx] = f2bf(in[idx]);
}

// ---- fp32 [K,N] -> bf16 WMMA-B-fragment tiled layout [K/32][Npad][2][16] ----
// element (k,n): kt=k/32, kr=k%32, hb=(kr/8)%2, e=kr%8+8*(kr/16) -> out[(((kt*Npad+n)*2)+hb)*16+e]
__global__ void k_convB(const float* __restrict__ in, unsigned short* __restrict__ out,
                        int K, int N, int Npad) {
    int idx = blockIdx.x * blockDim.x + threadIdx.x;
    int total = K * Npad;
    if (idx >= total) return;
    int k = idx / Npad, n = idx - k * Npad;
    int kt = k >> 5, kr = k & 31, hb = (kr >> 3) & 1, e = (kr & 7) | ((kr >> 4) << 3);
    unsigned short v = (n < N) ? f2bf(in[(size_t)k * N + n]) : (unsigned short)0;
    out[((((size_t)kt * Npad + n) * 2) + hb) * 16 + e] = v;
}

// ---- mean over L ------------------------------------------------------------
__global__ void k_mean(const float* __restrict__ img, float* __restrict__ avg) {
    int idx = blockIdx.x * blockDim.x + threadIdx.x;   // b*DIMG + d
    if (idx >= B_ * DIMG) return;
    int b = idx / DIMG, d = idx - b * DIMG;
    const float* p = img + (size_t)b * L_ * DIMG + d;
    float s = 0.f;
    for (int l = 0; l < L_; ++l) s += p[(size_t)l * DIMG];
    avg[idx] = s * (1.0f / (float)L_);
}

// ---- WMMA bf16 GEMM: D[M,N] = act(A[M,K] * B[K,N] + bias (+D)) --------------
// A: bf16 row-major (ABF) or fp32 row-major converted in-register (!ABF).
// B: bf16 in fragment-tiled layout (see k_convB). Wave computes 16x32 strip.
// K % 32 == 0, M % 16 == 0, N % 32 == 0 (Nbound limits stores/bias).
template<bool ABF>
__global__ void k_gemm(const void* __restrict__ Aptr, int lda,
                       const unsigned short* __restrict__ Bw,
                       const float* __restrict__ bias0, const float* __restrict__ bias1,
                       float* __restrict__ D, int ldd,
                       int N, int K, int Nbound, int act, int accum) {
    int wave  = threadIdx.x >> 5;
    int lane  = threadIdx.x & 31;
    int nt0   = (blockIdx.x * 8 + wave) * 2;
    int mtile = blockIdx.y;
    if (nt0 * 16 >= N) return;   // whole-wave uniform exit (EXEC stays all-ones)

    int l15  = lane & 15;
    int hb   = lane >> 4;        // K-half select (0: K 0-7,16-23 / 1: K 8-15,24-31)
    int lo   = hb << 3;          // A-frag K-half offset: 0 or 8
    int rA   = mtile * 16 + l15;
    int col0 = nt0 * 16 + l15;
    int col1 = col0 + 16;

    const unsigned short* Ab = (const unsigned short*)Aptr;
    const float*          Af = (const float*)Aptr;

    v8f acc0 = {}, acc1 = {};
    int KT = K >> 5;
    for (int kt = 0; kt < KT; ++kt) {
        int kb = kt << 5;
        Frag16 af;
        if (ABF) {
            const unsigned short* Ar = Ab + (size_t)rA * lda + kb;
#pragma unroll
            for (int p = 0; p < 8; ++p) {          // K-pairs contiguous -> b32 loads
                int k0 = (p < 4) ? (lo + 2 * p) : (16 + lo + 2 * (p - 4));
                af.u[p] = *(const unsigned int*)(Ar + k0);
            }
        } else {
            const float* Ar = Af + (size_t)rA * lda + kb;
#pragma unroll
            for (int e = 0; e < 16; ++e) {
                int vv = e >> 1, j = e & 1;
                int k = (vv < 4) ? (lo + vv * 2 + j) : (16 + lo + (vv - 4) * 2 + j);
                af.s[e] = f2bf(Ar[k]);
            }
        }
        // B fragments: contiguous 32B aligned chunks in tiled layout
        const v16bf b0 = *(const v16bf*)(Bw + ((((size_t)kt * N + col0) * 2) + hb) * 16);
        const v16bf b1 = *(const v16bf*)(Bw + ((((size_t)kt * N + col1) * 2) + hb) * 16);
        // prefetch next K-tile of B (global_prefetch_b8; speculative OOB is dropped)
        __builtin_prefetch(Bw + ((((size_t)(kt + 1) * N + col0) * 2) + hb) * 16, 0, 1);
        acc0 = __builtin_amdgcn_wmma_f32_16x16x32_bf16(
                   false, af.v, false, b0, (short)0, acc0, false, false);
        acc1 = __builtin_amdgcn_wmma_f32_16x16x32_bf16(
                   false, af.v, false, b1, (short)0, acc1, false, false);
    }

#pragma unroll
    for (int tile = 0; tile < 2; ++tile) {
        int n = (nt0 + tile) * 16 + l15;
        v8f a = tile ? acc1 : acc0;
        if (n < Nbound) {
#pragma unroll
            for (int r = 0; r < 8; ++r) {          // C/D: M = r + 8*hb
                int m = mtile * 16 + r + (hb << 3);
                size_t idx = (size_t)m * ldd + n;
                float v = a[r];
                if (bias0) v += bias0[n];
                if (bias1) v += bias1[n];
                if (accum) v += D[idx];
                if (act == 1)      v = tanhf(v);
                else if (act == 2) v = 1.0f / (1.0f + expf(-v));
                D[idx] = v;
            }
        }
    }
}

// ---- attention energy: e[b,l] = sum_k tanh(Ws+Uh)*Wv + bv -------------------
__global__ void k_att_e(const float* __restrict__ Ws, const float* __restrict__ Uh,
                        const float* __restrict__ Wv, const float* __restrict__ bv,
                        float* __restrict__ e) {
    int bl = blockIdx.x;                 // b*L_ + l
    int b  = bl / L_;
    __shared__ float red[256];
    const float* wsrow = Ws + (size_t)bl * RNN_;
    const float* uhrow = Uh + (size_t)b * RNN_;
    float s = 0.f;
    for (int k = threadIdx.x; k < RNN_; k += 256)
        s += tanhf(wsrow[k] + uhrow[k]) * Wv[k];
    red[threadIdx.x] = s;
    __syncthreads();
    for (int st = 128; st > 0; st >>= 1) {
        if (threadIdx.x < st) red[threadIdx.x] += red[threadIdx.x + st];
        __syncthreads();
    }
    if (threadIdx.x == 0) e[bl] = red[0] + bv[0];
}

// ---- softmax over L; writes alpha to ws and to d_out ------------------------
__global__ void k_softmax(const float* __restrict__ e, float* __restrict__ alpha,
                          float* __restrict__ alpha_out) {
    int b = blockIdx.x;
    __shared__ float red[256];
    float v = (threadIdx.x < L_) ? e[b * L_ + threadIdx.x] : -3.0e38f;
    red[threadIdx.x] = v;
    __syncthreads();
    for (int st = 128; st > 0; st >>= 1) {
        if (threadIdx.x < st) red[threadIdx.x] = fmaxf(red[threadIdx.x], red[threadIdx.x + st]);
        __syncthreads();
    }
    float mx = red[0];
    __syncthreads();
    float ex = (threadIdx.x < L_) ? expf(v - mx) : 0.f;
    red[threadIdx.x] = ex;
    __syncthreads();
    for (int st = 128; st > 0; st >>= 1) {
        if (threadIdx.x < st) red[threadIdx.x] += red[threadIdx.x + st];
        __syncthreads();
    }
    float inv = 1.0f / red[0];
    if (threadIdx.x < L_) {
        float a = ex * inv;
        alpha[b * L_ + threadIdx.x] = a;
        alpha_out[(size_t)b * (TSTEPS * L_) + threadIdx.x] = a;  // base incl. t*L_
    }
}

// ---- context[b,d] = sum_l img[b,l,d]*alpha[b,l] -----------------------------
__global__ void k_ctx(const float* __restrict__ img, const float* __restrict__ alpha,
                      float* __restrict__ ctx) {
    int idx = blockIdx.x * blockDim.x + threadIdx.x;
    if (idx >= B_ * DIMG) return;
    int b = idx / DIMG, d = idx - b * DIMG;
    const float* p = img + (size_t)b * L_ * DIMG + d;
    const float* a = alpha + b * L_;
    float s = 0.f;
    for (int l = 0; l < L_; ++l) s += p[(size_t)l * DIMG] * a[l];
    ctx[idx] = s;
}

// ---- x = [emb(target_t), gate*context] --------------------------------------
__global__ void k_buildx(const float* __restrict__ emb, const int* __restrict__ captions,
                         const float* __restrict__ gate, const float* __restrict__ ctx,
                         float* __restrict__ x, int t) {
    int idx = blockIdx.x * blockDim.x + threadIdx.x;
    const int XW = EMB_ + DIMG;
    if (idx >= B_ * XW) return;
    int b = idx / XW, d = idx - b * XW;
    float v;
    if (d < EMB_) {
        int tok = (t == 0) ? START_TOK : captions[b * 16 + (t - 1)];
        v = emb[(size_t)tok * EMB_ + d];
    } else {
        int dd = d - EMB_;
        v = gate[b * DIMG + dd] * ctx[b * DIMG + dd];
    }
    x[idx] = v;
}

// ---- LSTM pointwise update --------------------------------------------------
__global__ void k_lstm(const float* __restrict__ gates, float* __restrict__ h,
                       float* __restrict__ c) {
    int idx = blockIdx.x * blockDim.x + threadIdx.x;
    if (idx >= B_ * RNN_) return;
    int b = idx / RNN_, j = idx - b * RNN_;
    const float* g = gates + (size_t)b * 4 * RNN_;
    float gi = g[j], gf = g[RNN_ + j], gg = g[2 * RNN_ + j], go = g[3 * RNN_ + j];
    float si = 1.f / (1.f + expf(-gi));
    float sf = 1.f / (1.f + expf(-gf));
    float so = 1.f / (1.f + expf(-go));
    float cn = sf * c[idx] + si * tanhf(gg);
    float hn = so * tanhf(cn);
    c[idx] = cn;
    h[idx] = hn;
}

extern "C" void kernel_launch(void* const* d_in, const int* in_sizes, int n_in,
                              void* d_out, int out_size, void* d_ws, size_t ws_size,
                              hipStream_t stream) {
    (void)in_sizes; (void)n_in; (void)out_size; (void)ws_size;

    const float* img      = (const float*)d_in[0];
    const int*   captions = (const int*)  d_in[1];
    const float* emb      = (const float*)d_in[2];
    const float* Wih      = (const float*)d_in[3];
    const float* Whh      = (const float*)d_in[4];
    const float* bih      = (const float*)d_in[5];
    const float* bhh      = (const float*)d_in[6];
    const float* Wh0      = (const float*)d_in[7];
    const float* bh0      = (const float*)d_in[8];
    const float* Wc0      = (const float*)d_in[9];
    const float* bc0      = (const float*)d_in[10];
    const float* Wfb      = (const float*)d_in[11];
    const float* bfb      = (const float*)d_in[12];
    const float* Wout     = (const float*)d_in[13];
    const float* bout     = (const float*)d_in[14];
    const float* WU       = (const float*)d_in[15];
    const float* bU       = (const float*)d_in[16];
    const float* WW       = (const float*)d_in[17];
    const float* bW       = (const float*)d_in[18];
    const float* Wv       = (const float*)d_in[19];
    const float* bv       = (const float*)d_in[20];
    float* out = (float*)d_out;

    // ---- workspace carve-out (~123 MB total) ----
    char* p = (char*)d_ws;
    auto alloc_us = [&](size_t n) { unsigned short* r = (unsigned short*)p;
                                    p += ((n * 2 + 255) / 256) * 256; return r; };
    auto alloc_f  = [&](size_t n) { float* r = (float*)p;
                                    p += ((n * 4 + 255) / 256) * 256; return r; };
    unsigned short* WihB  = alloc_us((size_t)2560 * 4096);
    unsigned short* WhhB  = alloc_us((size_t)1024 * 4096);
    unsigned short* WoutB = alloc_us((size_t)1024 * VOUT_PAD);
    unsigned short* WfbB  = alloc_us((size_t)1024 * 2048);
    unsigned short* WUB   = alloc_us((size_t)1024 * 1024);
    unsigned short* WWB   = alloc_us((size_t)2048 * 1024);
    unsigned short* Wh0B  = alloc_us((size_t)2048 * 1024);
    unsigned short* Wc0B  = alloc_us((size_t)2048 * 1024);
    unsigned short* avgB  = alloc_us((size_t)B_ * DIMG);
    unsigned short* hB    = alloc_us((size_t)B_ * RNN_);
    unsigned short* xB    = alloc_us((size_t)B_ * (EMB_ + DIMG));
    float* Ws    = alloc_f((size_t)B_ * L_ * RNN_);   // 51.4 MB, L2-resident
    float* avg   = alloc_f((size_t)B_ * DIMG);
    float* h     = alloc_f((size_t)B_ * RNN_);
    float* c     = alloc_f((size_t)B_ * RNN_);
    float* Uh    = alloc_f((size_t)B_ * RNN_);
    float* e     = alloc_f((size_t)B_ * L_);
    float* alpha = alloc_f((size_t)B_ * L_);
    float* ctx   = alloc_f((size_t)B_ * DIMG);
    float* gate  = alloc_f((size_t)B_ * DIMG);
    float* x     = alloc_f((size_t)B_ * (EMB_ + DIMG));
    float* gates = alloc_f((size_t)B_ * 4 * RNN_);

    auto convB = [&](const float* in, unsigned short* o, int K, int n, int npad) {
        int tot = K * npad;
        k_convB<<<(tot + 255) / 256, 256, 0, stream>>>(in, o, K, n, npad);
    };
    auto convA = [&](const float* in, unsigned short* o, int tot) {
        k_convA<<<(tot + 255) / 256, 256, 0, stream>>>(in, o, tot);
    };
    auto gemmB = [&](const unsigned short* A, int lda, const unsigned short* Bw,
                     const float* b0, const float* b1, float* Dp, int ldd,
                     int M, int N, int K, int Nbound, int act, int accum) {
        dim3 grid((N / 32 + 7) / 8, M / 16);
        k_gemm<true><<<grid, 256, 0, stream>>>(A, lda, Bw, b0, b1, Dp, ldd,
                                               N, K, Nbound, act, accum);
    };
    auto gemmF = [&](const float* A, int lda, const unsigned short* Bw,
                     const float* b0, const float* b1, float* Dp, int ldd,
                     int M, int N, int K, int Nbound, int act, int accum) {
        dim3 grid((N / 32 + 7) / 8, M / 16);
        k_gemm<false><<<grid, 256, 0, stream>>>(A, lda, Bw, b0, b1, Dp, ldd,
                                                N, K, Nbound, act, accum);
    };

    // ---- one-time (per launch) weight converts into WMMA-B tiled layout ----
    convB(Wih,  WihB,  2560, 4096, 4096);
    convB(Whh,  WhhB,  1024, 4096, 4096);
    convB(Wout, WoutB, 1024, VOUT, VOUT_PAD);
    convB(Wfb,  WfbB,  1024, 2048, 2048);
    convB(WU,   WUB,   1024, 1024, 1024);
    convB(WW,   WWB,   2048, 1024, 1024);
    convB(Wh0,  Wh0B,  2048, 1024, 1024);
    convB(Wc0,  Wc0B,  2048, 1024, 1024);

    // ---- init state + timestep-independent attention projection ----
    k_mean<<<(B_ * DIMG + 255) / 256, 256, 0, stream>>>(img, avg);
    convA(avg, avgB, B_ * DIMG);
    gemmB(avgB, DIMG, Wh0B, bh0, nullptr, h, RNN_, B_, RNN_, DIMG, RNN_, 1, 0);
    gemmB(avgB, DIMG, Wc0B, bc0, nullptr, c, RNN_, B_, RNN_, DIMG, RNN_, 1, 0);
    // big one-shot GEMM: A stays fp32 (in-register convert) to avoid 51MB copy
    gemmF(img, DIMG, WWB, bW, nullptr, Ws, RNN_, B_ * L_, RNN_, DIMG, RNN_, 0, 0);
    convA(h, hB, B_ * RNN_);

    const size_t PREDS = (size_t)B_ * TSTEPS * VOUT;
    for (int t = 0; t < TSTEPS; ++t) {
        // U_h = h @ WU + bU
        gemmB(hB, RNN_, WUB, bU, nullptr, Uh, RNN_, B_, RNN_, RNN_, RNN_, 0, 0);
        // attention energies, softmax (alphas straight into d_out), context
        k_att_e<<<B_ * L_, 256, 0, stream>>>(Ws, Uh, Wv, bv, e);
        k_softmax<<<B_, 256, 0, stream>>>(e, alpha, out + PREDS + (size_t)t * L_);
        k_ctx<<<(B_ * DIMG + 255) / 256, 256, 0, stream>>>(img, alpha, ctx);
        // gate = sigmoid(h @ Wfb + bfb)
        gemmB(hB, RNN_, WfbB, bfb, nullptr, gate, DIMG, B_, DIMG, RNN_, DIMG, 2, 0);
        // x = [emb_t, gate*context] -> bf16
        k_buildx<<<(B_ * (EMB_ + DIMG) + 255) / 256, 256, 0, stream>>>(
            emb, captions, gate, ctx, x, t);
        convA(x, xB, B_ * (EMB_ + DIMG));
        // gates = x @ Wih + (bih + bhh); gates += h @ Whh
        gemmB(xB, EMB_ + DIMG, WihB, bih, bhh, gates, 4096, B_, 4096, EMB_ + DIMG, 4096, 0, 0);
        gemmB(hB, RNN_,        WhhB, nullptr, nullptr, gates, 4096, B_, 4096, RNN_, 4096, 0, 1);
        // LSTM pointwise (updates h, c), re-convert h for next GEMMs
        k_lstm<<<(B_ * RNN_ + 255) / 256, 256, 0, stream>>>(gates, h, c);
        convA(h, hB, B_ * RNN_);
        // preds[:, t, :] = h_new @ Wout + bout (N padded to 10016, bounded 10003)
        gemmB(hB, RNN_, WoutB, bout, nullptr,
              out + (size_t)t * VOUT, TSTEPS * VOUT,
              B_, VOUT_PAD, RNN_, VOUT, 0, 0);
    }
}
